// S4DEnc_17197049053648
// MI455X (gfx1250) — matrix operand order repaired and checked by reading, not compiled, not run on hardware
//
#include <hip/hip_runtime.h>
#include <hip/hip_bf16.h>

// ---------------- CDNA5 WMMA types ----------------
typedef __attribute__((ext_vector_type(16))) __bf16         v16bf;
typedef __attribute__((ext_vector_type(8)))  float          v8f;
typedef __attribute__((ext_vector_type(16))) unsigned short v16us;
typedef __attribute__((ext_vector_type(8)))  unsigned short v8us;

#define DEVFN static __device__ __forceinline__

DEVFN unsigned short f2bf(float f) {
  unsigned int u = __float_as_uint(f);
  u += 0x7FFFu + ((u >> 16) & 1u);   // round-to-nearest-even
  return (unsigned short)(u >> 16);
}
DEVFN float geluf(float x) { return 0.5f * x * (1.0f + erff(x * 0.7071067811865475f)); }

// A fragment (16x32 bf16 tile in LDS, row stride 32):
// lane m=lane&15 holds row m; kh=lane>>4 selects K {kh*16..+15}
DEVFN v16bf load_afrag(const unsigned short* As, int lane) {
  int m = lane & 15, kh = lane >> 4;
  const unsigned short* p = As + m * 32 + kh * 16;
  v8us lo = *(const v8us*)(p);
  v8us hi = *(const v8us*)(p + 8);
  v16us t = __builtin_shufflevector(lo, hi, 0,1,2,3,4,5,6,7,8,9,10,11,12,13,14,15);
  return __builtin_bit_cast(v16bf, t);
}
// B fragment from pre-transposed Bt[N][K] bf16: lane n=lane&15, K-half by lane>>4;
// 16 contiguous bf16 = one 32B load per lane.
DEVFN v16bf load_bfrag(const unsigned short* Bt, int nbase, int K, int k0, int lane) {
  int n  = nbase + (lane & 15);
  int kb = k0 + (lane >> 4) * 16;
  v16us t = *(const v16us*)(Bt + (long)n * K + kb);
  return __builtin_bit_cast(v16bf, t);
}
DEVFN v8f wmma_bf16(v16bf a, v16bf b, v8f c) {
  return __builtin_amdgcn_wmma_f32_16x16x32_bf16(false, a, false, b, (short)0, c, false, false);
}

// ---------------- small helper kernels ----------------
__global__ void zero_kernel(float* p, int n) {
  int i = blockIdx.x * blockDim.x + threadIdx.x;
  if (i < n) p[i] = 0.0f;
}

// f32 -> bf16 convert, optional transpose (src [R][C] -> dst [C][R])
__global__ void cvt_bf16_kernel(const float* __restrict__ src, unsigned short* __restrict__ dst,
                                int R, int C, int transpose) {
  int i = blockIdx.x * blockDim.x + threadIdx.x;
  if (i >= R * C) return;
  int r = i / C, c = i % C;
  if (transpose) dst[c * R + r] = f2bf(src[i]);
  else           dst[i]         = f2bf(src[i]);
}

// S4D discretization: decay = exp(dt*A), Ck2 = 2*C*(exp(dt*A)-1)/A  (complex, per h,n)
__global__ void s4_prep_kernel(const float* __restrict__ log_dt, const float* __restrict__ Cre,
                               const float* __restrict__ Cim, const float* __restrict__ logAre,
                               const float* __restrict__ Aim,
                               float* __restrict__ decr, float* __restrict__ deci,
                               float* __restrict__ ckr,  float* __restrict__ cki, int total) {
  int i = blockIdx.x * blockDim.x + threadIdx.x;
  if (i >= total) return;
  int h = i >> 6;                            // N = 64 modes
  float dt  = expf(log_dt[h]);
  float ar  = -expf(logAre[i]);
  float ai  = Aim[i];
  float er  = expf(ar * dt);
  float dr  = er * cosf(ai * dt);
  float di  = er * sinf(ai * dt);
  float e1r = dr - 1.0f, e1i = di;
  float inv = 1.0f / (ar * ar + ai * ai);
  float qr  = (e1r * ar + e1i * ai) * inv;   // (exp(dtA)-1)/A = (exp-1)*conj(A)/|A|^2
  float qi  = (e1i * ar - e1r * ai) * inv;
  float cr  = Cre[i], ci = Cim[i];
  decr[i] = dr; deci[i] = di;
  ckr[i] = 2.0f * (cr * qr - ci * qi);
  cki[i] = 2.0f * (cr * qi + ci * qr);
}

// sents = acc/L + d2b  (pooled word output, fed to sentence encoder)
__global__ void fixup_kernel(const float* __restrict__ acc, const float* __restrict__ d2b,
                             float* __restrict__ out, int n, float invL) {
  int i = blockIdx.x * blockDim.x + threadIdx.x;
  if (i < n) out[i] = acc[i] * invL + d2b[i & 255];
}

// ---------------- fused gather + encoder GEMM (WMMA, double-buffered LDS) ----------------
// u[seq][h][l] = sum_e rows[gather(row)][e] * encW[e][h] + encb[h];  H==128, 8 waves/WG
__global__ __launch_bounds__(256) void enc_gemm_kernel(
    const float* __restrict__ rows, const int* __restrict__ ids,
    const unsigned short* __restrict__ Wt /*[H][E] bf16*/, const float* __restrict__ bias,
    float* __restrict__ u, int E, int lsh) {
  __shared__ __align__(32) unsigned short As[2][512];
  const int H = 128;
  int tid = threadIdx.x, lane = tid & 31, w = tid >> 5;
  int row0 = blockIdx.x * 16;
  // fixed staging slot: thread -> (row sr, col pair sc); gather resolved ONCE
  int sr = tid >> 4, sc = (tid & 15) * 2;
  int grow = row0 + sr;
  int src = ids ? ids[grow] : grow;
  const float* rowp = rows + (long)src * E;
  unsigned int* d0 = (unsigned int*)&As[0][sr * 32 + sc];
  unsigned int* d1 = (unsigned int*)&As[1][sr * 32 + sc];
  auto stage = [&](int buf, int k0) {
    float2 v = *(const float2*)(rowp + k0 + sc);
    unsigned int pk = (unsigned int)f2bf(v.x) | ((unsigned int)f2bf(v.y) << 16);
    *(buf ? d1 : d0) = pk;
    __builtin_prefetch(rowp + k0 + 64, 0, 0);   // next K-chunk of gathered row
  };
  stage(0, 0);
  v8f acc = {0, 0, 0, 0, 0, 0, 0, 0};
  int buf = 0;
  for (int k0 = 0; k0 < E; k0 += 32) {
    __syncthreads();
    if (k0 + 32 < E) stage(buf ^ 1, k0 + 32);   // overlap next stage with WMMA
    v16bf a = load_afrag(As[buf], lane);
    v16bf b = load_bfrag(Wt, w * 16, E, k0, lane);
    acc = wmma_bf16(a, b, acc);
    buf ^= 1;
  }
  int h = w * 16 + (lane & 15);
  int kh = lane >> 4;
  int lmask = (1 << lsh) - 1;
  float bv = bias[h];
  #pragma unroll
  for (int r = 0; r < 8; ++r) {
    int row = row0 + r + 8 * kh;
    int seq = row >> lsh, l = row & lmask;
    u[((long)(seq << 7) + h) * (lmask + 1) + l] = acc[r] + bv;
  }
}

// ---------------- S4D recurrent scan (== truncated-FFT conv) + D-skip + GELU ----------------
// one wave per (seq,h); lane owns modes {lane, lane+32}; butterfly reduce per step
__global__ __launch_bounds__(256) void s4_scan_kernel(
    const float* __restrict__ u, const float* __restrict__ decr, const float* __restrict__ deci,
    const float* __restrict__ ckr, const float* __restrict__ cki, const float* __restrict__ Dp,
    unsigned short* __restrict__ yact, int nwaves, int L) {
  const int H = 128;
  int wid  = blockIdx.x * (blockDim.x >> 5) + (threadIdx.x >> 5);
  int lane = threadIdx.x & 31;
  if (wid >= nwaves) return;
  int seq = wid >> 7, h = wid & 127;
  int i0 = (h << 6) + lane, i1 = i0 + 32;
  float dr0 = decr[i0], di0 = deci[i0], cr0 = ckr[i0], ci0 = cki[i0];
  float dr1 = decr[i1], di1 = deci[i1], cr1 = ckr[i1], ci1 = cki[i1];
  float Dh = Dp[h];
  float s0r = 0, s0i = 0, s1r = 0, s1i = 0;
  const float* up = u + ((long)seq * H + h) * L;
  unsigned short* yp = yact + (long)seq * L * H + h;
  for (int l = 0; l < L; ++l) {
    float uv = up[l];                          // wave-uniform broadcast load
    float t0r = dr0 * s0r - di0 * s0i + uv;
    float t0i = dr0 * s0i + di0 * s0r;
    float t1r = dr1 * s1r - di1 * s1i + uv;
    float t1i = dr1 * s1i + di1 * s1r;
    s0r = t0r; s0i = t0i; s1r = t1r; s1i = t1i;
    float part = cr0 * s0r - ci0 * s0i + cr1 * s1r - ci1 * s1i;
    #pragma unroll
    for (int m = 16; m >= 1; m >>= 1) part += __shfl_xor(part, m, 32);
    if (lane == 0) yp[(long)l * H] = f2bf(geluf(part + Dh * uv));
  }
}

// ---------------- fused 1x1-conv GEMM + GLU + residual + LayerNorm (WMMA) ----------------
// wave w accumulates output channels {16w..} and {128+16w..}; GLU pairs stay in-wave.
__global__ __launch_bounds__(256) void conv_glu_ln_kernel(
    const unsigned short* __restrict__ yact /*[M][128] bf16*/,
    const unsigned short* __restrict__ Wt /*[256][128] bf16*/, const float* __restrict__ convb,
    const float* __restrict__ u, const float* __restrict__ lng, const float* __restrict__ lnb,
    unsigned short* __restrict__ xn, int lsh) {
  __shared__ __align__(32) unsigned short As[2][512];
  __shared__ float zs[16][128];
  const int H = 128;
  int tid = threadIdx.x, lane = tid & 31, w = tid >> 5;
  int row0 = blockIdx.x * 16;
  int sr = tid >> 4, sc = (tid & 15) * 2;
  const unsigned short* ap = yact + (long)(row0 + sr) * H + sc;
  unsigned int* d0 = (unsigned int*)&As[0][sr * 32 + sc];
  unsigned int* d1 = (unsigned int*)&As[1][sr * 32 + sc];
  auto stage = [&](int buf, int k0) { *(buf ? d1 : d0) = *(const unsigned int*)(ap + k0); };
  stage(0, 0);
  v8f ca = {0,0,0,0,0,0,0,0}, cb2 = {0,0,0,0,0,0,0,0};
  int buf = 0;
  for (int k0 = 0; k0 < H; k0 += 32) {
    __syncthreads();
    if (k0 + 32 < H) stage(buf ^ 1, k0 + 32);
    v16bf a  = load_afrag(As[buf], lane);
    v16bf b0 = load_bfrag(Wt, w * 16,     H, k0, lane);
    v16bf b1 = load_bfrag(Wt, H + w * 16, H, k0, lane);
    ca  = wmma_bf16(a, b0, ca);
    cb2 = wmma_bf16(a, b1, cb2);
    buf ^= 1;
  }
  int h = w * 16 + (lane & 15);
  int kh = lane >> 4;
  int lmask = (1 << lsh) - 1;
  float ba = convb[h], bb = convb[h + 128];
  #pragma unroll
  for (int r = 0; r < 8; ++r) {
    int ml = r + 8 * kh;
    int row = row0 + ml;
    int seq = row >> lsh, l = row & lmask;
    float av = ca[r] + ba, bv = cb2[r] + bb;
    float z = av * (1.0f / (1.0f + expf(-bv)));       // GLU
    z += u[((long)(seq << 7) + h) * (lmask + 1) + l]; // residual (pre-S4 encoder output)
    zs[ml][h] = z;
  }
  __syncthreads();
  for (int rr = w; rr < 16; rr += 8) {                // LayerNorm over H=128 per row
    float x[4], s = 0.0f, ss = 0.0f;
    #pragma unroll
    for (int t = 0; t < 4; ++t) { x[t] = zs[rr][lane + 32 * t]; s += x[t]; ss += x[t] * x[t]; }
    #pragma unroll
    for (int m = 16; m >= 1; m >>= 1) { s += __shfl_xor(s, m, 32); ss += __shfl_xor(ss, m, 32); }
    float mu  = s * (1.0f / 128.0f);
    float var = ss * (1.0f / 128.0f) - mu * mu;       // biased var (matches jnp.var)
    float rs  = rsqrtf(var + 1e-5f);
    int row = row0 + rr;
    #pragma unroll
    for (int t = 0; t < 4; ++t) {
      int hh = lane + 32 * t;
      xn[(long)row * H + hh] = f2bf((x[t] - mu) * rs * lng[hh] + lnb[hh]);
    }
  }
}

// ---------------- FFN GEMMs (WMMA, N=256 dual-tile per wave) ----------------
// EPI 0: GELU -> bf16 store [M][256]   (d1)
// EPI 1: mean-pool accumulate: atomicAdd into acc[seq][256], no bias (d2, word)
// EPI 2: +bias -> f32 store [M][256]   (d2, sentence -> d_out)
template <int EPI>
__global__ __launch_bounds__(256) void ffn_gemm_kernel(
    const unsigned short* __restrict__ A /*[M][K] bf16*/,
    const unsigned short* __restrict__ Bt /*[256][K] bf16*/,
    const float* __restrict__ bias, void* __restrict__ outp, int K, int lsh) {
  __shared__ __align__(32) unsigned short As[2][512];
  int tid = threadIdx.x, lane = tid & 31, w = tid >> 5;
  int row0 = blockIdx.x * 16;
  int sr = tid >> 4, sc = (tid & 15) * 2;
  const unsigned short* ap = A + (long)(row0 + sr) * K + sc;
  unsigned int* d0 = (unsigned int*)&As[0][sr * 32 + sc];
  unsigned int* d1 = (unsigned int*)&As[1][sr * 32 + sc];
  auto stage = [&](int buf, int k0) { *(buf ? d1 : d0) = *(const unsigned int*)(ap + k0); };
  stage(0, 0);
  v8f c0 = {0,0,0,0,0,0,0,0}, c1 = {0,0,0,0,0,0,0,0};
  int buf = 0;
  for (int k0 = 0; k0 < K; k0 += 32) {
    __syncthreads();
    if (k0 + 32 < K) stage(buf ^ 1, k0 + 32);
    v16bf a  = load_afrag(As[buf], lane);
    v16bf b0 = load_bfrag(Bt, w * 16,       K, k0, lane);
    v16bf b1 = load_bfrag(Bt, 128 + w * 16, K, k0, lane);
    c0 = wmma_bf16(a, b0, c0);
    c1 = wmma_bf16(a, b1, c1);
    buf ^= 1;
  }
  int o0 = w * 16 + (lane & 15), o1 = o0 + 128;
  int kh = lane >> 4;
  if (EPI == 0) {
    unsigned short* out = (unsigned short*)outp;
    float b0v = bias[o0], b1v = bias[o1];
    #pragma unroll
    for (int r = 0; r < 8; ++r) {
      long row = row0 + r + 8 * kh;
      out[row * 256 + o0] = f2bf(geluf(c0[r] + b0v));
      out[row * 256 + o1] = f2bf(geluf(c1[r] + b1v));
    }
  } else if (EPI == 1) {
    float* out = (float*)outp;
    float s0 = 0, s1 = 0;
    #pragma unroll
    for (int r = 0; r < 8; ++r) { s0 += c0[r]; s1 += c1[r]; }
    s0 += __shfl_xor(s0, 16, 32);
    s1 += __shfl_xor(s1, 16, 32);
    if (lane < 16) {
      int seq = row0 >> lsh;                  // 16-row tile never straddles a sequence
      atomicAdd(out + (long)seq * 256 + w * 16 + lane, s0);
      atomicAdd(out + (long)seq * 256 + 128 + w * 16 + lane, s1);
    }
  } else {
    float* out = (float*)outp;
    float b0v = bias[o0], b1v = bias[o1];
    #pragma unroll
    for (int r = 0; r < 8; ++r) {
      long row = row0 + r + 8 * kh;
      out[row * 256 + o0] = c0[r] + b0v;
      out[row * 256 + o1] = c1[r] + b1v;
    }
  }
}

// ---------------- host launcher ----------------
extern "C" void kernel_launch(void* const* d_in, const int* in_sizes, int n_in,
                              void* d_out, int out_size, void* d_ws, size_t ws_size,
                              hipStream_t stream) {
  (void)in_sizes; (void)n_in; (void)out_size; (void)ws_size;
  const int*   ids = (const int*)d_in[0];
  const float* emb = (const float*)d_in[1];
  const float* wp[16]; const float* sp[16];
  for (int i = 0; i < 16; ++i) { wp[i] = (const float*)d_in[2 + i]; sp[i] = (const float*)d_in[18 + i]; }
  enum { I_ENCW = 0, I_ENCB, I_LOGDT, I_CRE, I_CIM, I_LOGARE, I_AIM, I_D,
         I_CONVW, I_CONVB, I_LNG, I_LNB, I_D1W, I_D1B, I_D2W, I_D2B };

  const int  H = 128, E = 256;
  const int  NSEQ_W = 512, L_W = 512, LSH_W = 9; const long M_W = (long)NSEQ_W * L_W;  // 262144
  const int  NSEQ_S = 8,   L_S = 64,  LSH_S = 6; const long M_S = (long)NSEQ_S * L_S;  // 512

  char* ws = (char*)d_ws;
  float*          u_w    = (float*)(ws + 0);                          // 134,217,728 B
  unsigned short* yact_w = (unsigned short*)(ws + 134217728);         //  67,108,864 B
  unsigned short* xn_w   = (unsigned short*)(ws + 201326592);         //  67,108,864 B
  unsigned short* h1_w   = (unsigned short*)(ws + 0);                 // reuse (u dead)
  float*          sacc   = (float*)(ws + 268435456);                  //     524,288 B
  float*          sf     = (float*)(ws + 268435456 + 524288);         //     524,288 B
  size_t off = 268435456 + 1048576;
  float*          u_s    = (float*)(ws + off);          off += 262144;
  unsigned short* yact_s = (unsigned short*)(ws + off); off += 131072;
  unsigned short* xn_s   = (unsigned short*)(ws + off); off += 131072;
  unsigned short* h1_s   = (unsigned short*)(ws + off); off += 262144;
  float* wdec[4]; float* sdec[4];
  for (int j = 0; j < 4; ++j) { wdec[j] = (float*)(ws + off); off += 32768; }
  for (int j = 0; j < 4; ++j) { sdec[j] = (float*)(ws + off); off += 32768; }
  unsigned short* encWt_w  = (unsigned short*)(ws + off); off += 65536;
  unsigned short* convWt_w = (unsigned short*)(ws + off); off += 65536;
  unsigned short* d1Wt_w   = (unsigned short*)(ws + off); off += 65536;
  unsigned short* d2Wt_w   = (unsigned short*)(ws + off); off += 131072;
  unsigned short* encWt_s  = (unsigned short*)(ws + off); off += 65536;
  unsigned short* convWt_s = (unsigned short*)(ws + off); off += 65536;
  unsigned short* d1Wt_s   = (unsigned short*)(ws + off); off += 65536;
  unsigned short* d2Wt_s   = (unsigned short*)(ws + off); off += 131072;

  // --- parameter prep (S4D discretization + bf16 weight transposes) ---
  s4_prep_kernel<<<32, 256, 0, stream>>>(wp[I_LOGDT], wp[I_CRE], wp[I_CIM], wp[I_LOGARE], wp[I_AIM],
                                         wdec[0], wdec[1], wdec[2], wdec[3], 8192);
  s4_prep_kernel<<<32, 256, 0, stream>>>(sp[I_LOGDT], sp[I_CRE], sp[I_CIM], sp[I_LOGARE], sp[I_AIM],
                                         sdec[0], sdec[1], sdec[2], sdec[3], 8192);
  cvt_bf16_kernel<<<128, 256, 0, stream>>>(wp[I_ENCW],  encWt_w,  256, 128, 1);
  cvt_bf16_kernel<<<128, 256, 0, stream>>>(wp[I_CONVW], convWt_w, 256, 128, 0);
  cvt_bf16_kernel<<<128, 256, 0, stream>>>(wp[I_D1W],   d1Wt_w,   128, 256, 1);
  cvt_bf16_kernel<<<256, 256, 0, stream>>>(wp[I_D2W],   d2Wt_w,   256, 256, 1);
  cvt_bf16_kernel<<<128, 256, 0, stream>>>(sp[I_ENCW],  encWt_s,  256, 128, 1);
  cvt_bf16_kernel<<<128, 256, 0, stream>>>(sp[I_CONVW], convWt_s, 256, 128, 0);
  cvt_bf16_kernel<<<128, 256, 0, stream>>>(sp[I_D1W],   d1Wt_s,   128, 256, 1);
  cvt_bf16_kernel<<<256, 256, 0, stream>>>(sp[I_D2W],   d2Wt_s,   256, 256, 1);
  zero_kernel<<<512, 256, 0, stream>>>(sacc, NSEQ_W * 256);

  // --- word-level S4 model (pooling=True) ---
  enc_gemm_kernel<<<(int)(M_W / 16), 256, 0, stream>>>(emb, ids, encWt_w, wp[I_ENCB], u_w, E, LSH_W);
  s4_scan_kernel<<<(NSEQ_W * H) / 8, 256, 0, stream>>>(u_w, wdec[0], wdec[1], wdec[2], wdec[3],
                                                       wp[I_D], yact_w, NSEQ_W * H, L_W);
  conv_glu_ln_kernel<<<(int)(M_W / 16), 256, 0, stream>>>(yact_w, convWt_w, wp[I_CONVB], u_w,
                                                          wp[I_LNG], wp[I_LNB], xn_w, LSH_W);
  ffn_gemm_kernel<0><<<(int)(M_W / 16), 256, 0, stream>>>(xn_w, d1Wt_w, wp[I_D1B], h1_w, 128, LSH_W);
  ffn_gemm_kernel<1><<<(int)(M_W / 16), 256, 0, stream>>>(h1_w, d2Wt_w, nullptr, sacc, 256, LSH_W);
  fixup_kernel<<<512, 256, 0, stream>>>(sacc, wp[I_D2B], sf, NSEQ_W * 256, 1.0f / (float)L_W);

  // --- sentence-level S4 model (pooling=False) ---
  enc_gemm_kernel<<<(int)(M_S / 16), 256, 0, stream>>>(sf, nullptr, encWt_s, sp[I_ENCB], u_s, E, LSH_S);
  s4_scan_kernel<<<(NSEQ_S * H) / 8, 256, 0, stream>>>(u_s, sdec[0], sdec[1], sdec[2], sdec[3],
                                                       sp[I_D], yact_s, NSEQ_S * H, L_S);
  conv_glu_ln_kernel<<<(int)(M_S / 16), 256, 0, stream>>>(yact_s, convWt_s, sp[I_CONVB], u_s,
                                                          sp[I_LNG], sp[I_LNB], xn_s, LSH_S);
  ffn_gemm_kernel<0><<<(int)(M_S / 16), 256, 0, stream>>>(xn_s, d1Wt_s, sp[I_D1B], h1_s, 128, LSH_S);
  ffn_gemm_kernel<2><<<(int)(M_S / 16), 256, 0, stream>>>(h1_s, d2Wt_s, sp[I_D2B], d_out, 256, LSH_S);
}
